// GraphRNN_7413113553638
// MI455X (gfx1250) — matrix operand order repaired and checked by reading, not compiled, not run on hardware
//
#include <hip/hip_runtime.h>
#include <hip/hip_bf16.h>

#define N_SAMPLES 64
#define N_NODES   2000
#define N_TOTAL   128000
#define N_EDGES   2048000
#define HDIM      64

typedef __attribute__((ext_vector_type(16))) __bf16 v16bf;
typedef __attribute__((ext_vector_type(8)))  float  v8f;

union FragU { uint4 u[2]; v16bf v; };

__device__ __forceinline__ unsigned short f2bf(float x) {
    union { float f; unsigned u; } t; t.f = x;
    unsigned r = t.u + 0x7FFFu + ((t.u >> 16) & 1u);   // round-to-nearest-even
    return (unsigned short)(r >> 16);
}

// A fragment (16x32 bf16): lanes 0-15 hold M=lane, elems 0..7 -> K=kk*32+half*8+e,
// elems 8..15 -> +16. Two ds_load_b128 per fragment.
__device__ __forceinline__ v16bf frag_a(const unsigned short* row, int kk, int half) {
    const uint4* p = reinterpret_cast<const uint4*>(row + kk * 32 + half * 8);
    FragU f; f.u[0] = p[0]; f.u[1] = p[2];
    return f.v;
}
// B fragment (32x16 bf16): N=lane&15, elems 0..7 -> K=kk*32+half*8+e, elems 8..15 -> +16.
__device__ __forceinline__ v16bf frag_b(const unsigned short* row, int kk, int half) {
    const uint4* p = reinterpret_cast<const uint4*>(row + kk * 32 + half * 8);
    FragU f; f.u[0] = p[0]; f.u[1] = p[2];
    return f.v;
}

// ---------------- GCN kernels ----------------

__global__ void k_xw0(const float* __restrict__ x, const float* __restrict__ Wg0,
                      float* __restrict__ XW) {
    int idx = blockIdx.x * 256 + threadIdx.x;
    XW[idx] = x[idx >> 6] * Wg0[idx & 63];
}

__global__ void k_deg(const int* __restrict__ edst, const float* __restrict__ ew,
                      float* __restrict__ DEG) {
    int e = blockIdx.x * 256 + threadIdx.x;
    atomicAdd(&DEG[edst[e]], ew[e]);
}

__global__ void k_dis(float* __restrict__ DEG, float* __restrict__ DIS) {
    int i = blockIdx.x * 256 + threadIdx.x;
    float d = DEG[i] + 1.0f;
    DEG[i] = d;
    DIS[i] = rsqrtf(d);
}

__global__ void k_edge_agg(const int* __restrict__ esrc, const int* __restrict__ edst,
                           const float* __restrict__ ew, const float* __restrict__ DIS,
                           const float* __restrict__ XW, float* __restrict__ AGG) {
    int t = blockIdx.x * 256 + threadIdx.x;
    int e = t >> 6, c = t & 63;
    int s = esrc[e], d = edst[e];
    float norm = DIS[s] * ew[e] * DIS[d];
    atomicAdd(&AGG[d * 64 + c], norm * XW[s * 64 + c]);
}

__global__ void k_combine(const float* __restrict__ AGG, const float* __restrict__ XW,
                          const float* __restrict__ DEG, const float* __restrict__ bg,
                          float* __restrict__ Hout) {
    int idx = blockIdx.x * 256 + threadIdx.x;
    int i = idx >> 6, j = idx & 63;
    float v = AGG[idx] + XW[idx] / DEG[i] + bg[j];
    Hout[idx] = v > 0.f ? v : 0.f;
}

// XW = Hin(128000x64) @ W(64x64) via bf16 WMMA. Block=256thr(8 waves) does 32 rows.
__global__ void k_gemm64(const float* __restrict__ Hin, const float* __restrict__ W,
                         float* __restrict__ XW) {
    __shared__ unsigned short sWT[64 * 64]; // transposed: sWT[n][k] = W[k][n]
    __shared__ unsigned short sA[32 * 64];
    int tid = threadIdx.x;
    for (int idx = tid; idx < 4096; idx += 256) {
        int k = idx >> 6, j = idx & 63;
        sWT[j * 64 + k] = f2bf(W[idx]);
    }
    int r0 = blockIdx.x * 32;
    for (int idx = tid; idx < 2048; idx += 256)
        sA[idx] = f2bf(Hin[r0 * 64 + idx]);
    __syncthreads();
    int wave = tid >> 5, lane = tid & 31, half = (lane >> 4) & 1, l16 = lane & 15;
    int mt = wave >> 2, nt = wave & 3;
    const unsigned short* arow = sA  + (mt * 16 + l16) * 64;
    const unsigned short* brow = sWT + (nt * 16 + l16) * 64;
    v8f acc = {};
#pragma unroll
    for (int kk = 0; kk < 2; ++kk) {
        v16bf a = frag_a(arow, kk, half);
        v16bf b = frag_b(brow, kk, half);
        acc = __builtin_amdgcn_wmma_f32_16x16x32_bf16(false, a, false, b,
                                                      (short)0, acc, false, false);
    }
#pragma unroll
    for (int v = 0; v < 8; ++v) {
        int row = r0 + mt * 16 + half * 8 + v;
        int col = nt * 16 + l16;
        XW[row * 64 + col] = acc[v];
    }
}

// ---------------- LSTM scan (fused input+recurrent GEMM) ----------------
// grid.x = 2 (direction), 512 threads = 16 waves.
// sW  [256 x KDIM] bf16 : concat(Wih, Whh) row-major (gate-row n, k)
// sXH [ 64 x KDIM] bf16 : per-step [x_t | h] (rows = samples)
// Wave w: mt = w/4 (batch tile), nq = w%4 (column tile within each gate);
// computes tiles (mt, g*4+nq) for g=0..3 so gate math is register-local.
template <int IN_DIM>
__global__ void __launch_bounds__(512) k_lstm_scan(
    const float* __restrict__ IN, float* __restrict__ OUT,
    const float* __restrict__ Wih_base, int wih_dir_stride,
    const float* __restrict__ Whh_base,
    const float* __restrict__ bih_base, const float* __restrict__ bhh_base) {
    extern __shared__ unsigned short smem[];
    constexpr int KDIM   = IN_DIM + 64;
    constexpr int KTILES = KDIM / 32;
    constexpr int QROW   = IN_DIM / 4;     // float4 chunks per input row
    unsigned short* sW  = smem;            // 256 * KDIM
    unsigned short* sXH = smem + 256 * KDIM;

    int dir = blockIdx.x;
    int tid = threadIdx.x;
    const float* Wih = Wih_base + dir * wih_dir_stride;
    const float* Whh = Whh_base + dir * 256 * 64;
    const float* bih = bih_base + dir * 256;
    const float* bhh = bhh_base + dir * 256;

    for (int idx = tid; idx < 256 * KDIM; idx += 512) {
        int n = idx / KDIM, k = idx - n * KDIM;
        float w = (k < IN_DIM) ? Wih[n * IN_DIM + k] : Whh[n * 64 + (k - IN_DIM)];
        sW[idx] = f2bf(w);
    }
    for (int idx = tid; idx < 64 * 64; idx += 512)
        sXH[(idx >> 6) * KDIM + IN_DIM + (idx & 63)] = 0;

    int wave = tid >> 5, lane = tid & 31, half = (lane >> 4) & 1, l16 = lane & 15;
    int mt = wave >> 2, nq = wave & 3;
    const unsigned short* arow = sXH + (mt * 16 + l16) * KDIM;
    int n = nq * 16 + l16;

    float bsum[4];
#pragma unroll
    for (int g = 0; g < 4; ++g) bsum[g] = bih[g * 64 + n] + bhh[g * 64 + n];

    float cst[8];
#pragma unroll
    for (int v = 0; v < 8; ++v) cst[v] = 0.f;

    __syncthreads();

    for (int t = 0; t < N_NODES; ++t) {
        int tt = dir ? (N_NODES - 1 - t) : t;
        // stage x_t into sXH[:, 0:IN_DIM]: float4 loads -> packed 4xbf16 b64 stores
        for (int idx = tid; idx < 64 * QROW; idx += 512) {
            int s = idx / QROW;            // QROW is a power of two -> shift
            int q = idx - s * QROW;
            float4 xv = *reinterpret_cast<const float4*>(
                IN + (size_t)(s * N_NODES + tt) * IN_DIM + q * 4);
            unsigned long long pk =
                 (unsigned long long)f2bf(xv.x)
               | ((unsigned long long)f2bf(xv.y) << 16)
               | ((unsigned long long)f2bf(xv.z) << 32)
               | ((unsigned long long)f2bf(xv.w) << 48);
            *reinterpret_cast<unsigned long long*>(sXH + s * KDIM + q * 4) = pk;
        }
        // prefetch next timestep's rows (global_prefetch_b8), hides latency on
        // the serial chain; fire-and-forget, no counters.
        if (t + 1 < N_NODES && tid < 64) {
            int ttn = dir ? (N_NODES - 2 - t) : (t + 1);
            const float* p = IN + (size_t)(tid * N_NODES + ttn) * IN_DIM;
#pragma unroll
            for (int k = 0; k < IN_DIM; k += 32) __builtin_prefetch(p + k, 0, 1);
        }
        __syncthreads();   // x staged + prev-step h writes visible

        v8f z = {};
        v8f acc[4] = { z, z, z, z };
#pragma unroll
        for (int kk = 0; kk < KTILES; ++kk) {
            v16bf a = frag_a(arow, kk, half);
#pragma unroll
            for (int g = 0; g < 4; ++g) {
                const unsigned short* brow = sW + (g * 64 + n) * KDIM;
                v16bf b = frag_b(brow, kk, half);
                acc[g] = __builtin_amdgcn_wmma_f32_16x16x32_bf16(
                    false, a, false, b, (short)0, acc[g], false, false);
            }
        }
        __syncthreads();   // all sXH reads done before h overwrite

#pragma unroll
        for (int v = 0; v < 8; ++v) {
            float gi = acc[0][v] + bsum[0];
            float gf = acc[1][v] + bsum[1];
            float gg = acc[2][v] + bsum[2];
            float go = acc[3][v] + bsum[3];
            float si = 1.f / (1.f + __expf(-gi));
            float sf = 1.f / (1.f + __expf(-gf));
            float so = 1.f / (1.f + __expf(-go));
            float c  = sf * cst[v] + si * tanhf(gg);
            cst[v]   = c;
            float h  = so * tanhf(c);
            int m = mt * 16 + half * 8 + v;
            sXH[m * KDIM + IN_DIM + n] = f2bf(h);
            OUT[(size_t)(m * N_NODES + tt) * 128 + dir * 64 + n] = h;
        }
        // next iteration's post-stage barrier orders these h writes vs. reads
    }
}

// ---------------- Final FC ----------------
__global__ void k_fc(const float* __restrict__ SEQ, const float* __restrict__ Wfc,
                     const float* __restrict__ bfc, float* __restrict__ out) {
    int i = blockIdx.x * 256 + threadIdx.x;
    const float4* s = reinterpret_cast<const float4*>(SEQ + (size_t)i * 128);
    const float4* w = reinterpret_cast<const float4*>(Wfc);
    float acc = bfc[0];
#pragma unroll
    for (int j = 0; j < 32; ++j) {
        float4 a = s[j], b = w[j];
        acc += a.x * b.x + a.y * b.y + a.z * b.z + a.w * b.w;
    }
    out[i] = acc;
}

extern "C" void kernel_launch(void* const* d_in, const int* in_sizes, int n_in,
                              void* d_out, int out_size, void* d_ws, size_t ws_size,
                              hipStream_t stream) {
    (void)in_sizes; (void)n_in; (void)out_size; (void)ws_size;
    const float* x     = (const float*)d_in[0];
    const int*   esrc  = (const int*)  d_in[1];
    const int*   edst  = (const int*)  d_in[2];
    const float* ew    = (const float*)d_in[3];
    const float* Wg0   = (const float*)d_in[4];
    const float* Wg12  = (const float*)d_in[5];
    const float* bg    = (const float*)d_in[6];
    const float* Wih0  = (const float*)d_in[7];
    const float* Wih12 = (const float*)d_in[8];
    const float* Whh   = (const float*)d_in[9];
    const float* bih   = (const float*)d_in[10];
    const float* bhh   = (const float*)d_in[11];
    const float* Wfc   = (const float*)d_in[12];
    const float* bfc   = (const float*)d_in[13];

    float* ws   = (float*)d_ws;
    float* H0   = ws;              // 128000*64
    float* XW   = ws + 8192000;    // 128000*64
    float* AGG  = ws + 16384000;   // 128000*64
    float* DEG  = ws + 24576000;   // 128000
    float* DIS  = ws + 24704000;   // 128000
    float* SEQA = ws + 24832000;   // 128000*128
    float* SEQB = XW;              // overlays XW+AGG = 128000*128 (GCN done by then)

    // degrees (same for all GCN layers)
    hipMemsetAsync(DEG, 0, 128000 * sizeof(float), stream);
    k_deg<<<N_EDGES / 256, 256, 0, stream>>>(edst, ew, DEG);
    k_dis<<<N_TOTAL / 256, 256, 0, stream>>>(DEG, DIS);

    // GCN layer 0 (input dim 1 -> outer product)
    k_xw0<<<(N_TOTAL * 64) / 256, 256, 0, stream>>>(x, Wg0, XW);
    hipMemsetAsync(AGG, 0, (size_t)N_TOTAL * 64 * sizeof(float), stream);
    k_edge_agg<<<(N_EDGES * 64) / 256, 256, 0, stream>>>(esrc, edst, ew, DIS, XW, AGG);
    k_combine<<<(N_TOTAL * 64) / 256, 256, 0, stream>>>(AGG, XW, DEG, bg, H0);

    // GCN layers 1..2
    for (int l = 0; l < 2; ++l) {
        k_gemm64<<<N_TOTAL / 32, 256, 0, stream>>>(H0, Wg12 + l * 4096, XW);
        hipMemsetAsync(AGG, 0, (size_t)N_TOTAL * 64 * sizeof(float), stream);
        k_edge_agg<<<(N_EDGES * 64) / 256, 256, 0, stream>>>(esrc, edst, ew, DIS, XW, AGG);
        k_combine<<<(N_TOTAL * 64) / 256, 256, 0, stream>>>(AGG, XW, DEG, bg + (l + 1) * 64, H0);
    }

    // BiLSTM layers: dynamic LDS = (256+64)*KDIM bf16
    size_t sh0  = (size_t)(256 + 64) * 128 * sizeof(unsigned short); //  80 KB
    size_t sh12 = (size_t)(256 + 64) * 192 * sizeof(unsigned short); // 120 KB (<320KB WGP LDS)
    k_lstm_scan<64> <<<2, 512, sh0,  stream>>>(H0,   SEQA, Wih0,           256 * 64,  Whh,         bih,        bhh);
    k_lstm_scan<128><<<2, 512, sh12, stream>>>(SEQA, SEQB, Wih12,          256 * 128, Whh + 32768, bih + 512,  bhh + 512);
    k_lstm_scan<128><<<2, 512, sh12, stream>>>(SEQB, SEQA, Wih12 + 65536,  256 * 128, Whh + 65536, bih + 1024, bhh + 1024);

    // head
    k_fc<<<N_TOTAL / 256, 256, 0, stream>>>(SEQA, Wfc, bfc, (float*)d_out);
}
